// wide_basic_8117488189562
// MI455X (gfx1250) — matrix operand order repaired and checked by reading, not compiled, not run on hardware
//
#include <hip/hip_runtime.h>

// ---------------------------------------------------------------------------
// wide_basic residual block on gfx1250 via bf16 WMMA implicit GEMM.
//   B=64, C=32, H=W=128.
//   a1 = bf16 NHWC relu(bn1(x))                      (kernel 1)
//   a2 = bf16 NHWC relu(bn2(conv3x3(a1,w1)+b1))      (kernel 2, MODE=1)
//   out = f32 NCHW conv3x3(a2,w2)+b2 + x             (kernel 2, MODE=2)
// Conv tiling: wave = 32 pixels x 16 out-channels; the wave's 9 B-fragments
// (3x3 taps, 32K x 16N bf16 each) live in registers for the whole tile, so the
// inner loop is 2 ds_load_b128 (A) per v_wmma_f32_16x16x32_bf16.
// Workspace layout (bytes):
//   [0,            67108864)  a1  bf16 NHWC
//   [67108864,    134217728)  a2  bf16 NHWC
//   [134217728,   134236160)  wp1 packed weight fragments (9 taps x 2 ntiles)
//   [134236160,   134254592)  wp2 packed weight fragments
// ---------------------------------------------------------------------------

typedef __attribute__((ext_vector_type(16))) __bf16 v16bf;
typedef __attribute__((ext_vector_type(8)))  float  v8f;

struct alignas(16) U4 { unsigned x, y, z, w; };
struct alignas(8)  U2 { unsigned x, y; };
struct alignas(16) F4 { float x, y, z, w; };

union AFrag { U4 q[2]; v16bf v; };

__device__ inline unsigned short f2bf(float f) {
    unsigned u = __builtin_bit_cast(unsigned, f);
    // round-to-nearest-even truncation to bf16
    return (unsigned short)((u + 0x7FFFu + ((u >> 16) & 1u)) >> 16);
}

#define EPS 1e-5f

// ---------------------------------------------------------------------------
// Pack OIHW f32 weights into a contiguous-k per-lane B-fragment order (this code's choice, matching the A loads) for
// v_wmma_f32_16x16x32_bf16:  B is 32(K) x 16(N); lanes 0-15 hold K=0..15,
// lanes 16-31 hold K=16..31, column N = lane&15, 2 bf16 per dword.
// Fragment index = tap*2 + ntile; each fragment: 32 lanes x 16 bf16 (32B).
// ---------------------------------------------------------------------------
__global__ void pack_weights_kernel(const float* __restrict__ w,
                                    unsigned short* __restrict__ wp) {
    int tapnt = blockIdx.x;          // 0..17
    int tap   = tapnt >> 1;
    int nt    = tapnt & 1;
    int kh    = tap / 3, kw = tap % 3;
    int lane  = threadIdx.x;         // 0..31
    int n     = nt * 16 + (lane & 15);
    int kbase = (lane >> 4) * 16;
    unsigned short* dst = wp + ((size_t)tapnt * 32 + lane) * 16;
#pragma unroll
    for (int i = 0; i < 16; ++i) {
        int k = kbase + i;
        dst[i] = f2bf(w[((n * 32 + k) * 3 + kh) * 3 + kw]);
    }
}

// ---------------------------------------------------------------------------
// Fused BN1 + ReLU + f32->bf16, NCHW -> NHWC transpose through LDS.
// One block: 32 pixels (one row segment) x 32 channels.
// ---------------------------------------------------------------------------
__global__ __launch_bounds__(256)
void bn_relu_pack_kernel(const float* __restrict__ x,
                         const float* __restrict__ g,  const float* __restrict__ be,
                         const float* __restrict__ mn, const float* __restrict__ vr,
                         unsigned short* __restrict__ a1) {
    __shared__ unsigned short tile[32 * 32];
    int blk = blockIdx.x;            // B*H*4 = 32768
    int b   = blk >> 9;              // / (128*4)
    int rem = blk & 511;
    int h   = rem >> 2;
    int w0  = (rem & 3) * 32;
    int t   = threadIdx.x;

    int c  = t >> 3;
    int wl = (t & 7) * 4;
    float sc = g[c] * rsqrtf(vr[c] + EPS);
    float sh = be[c] - mn[c] * sc;
    const F4 v = *reinterpret_cast<const F4*>(
        x + (((size_t)b * 32 + c) * 128 + h) * 128 + w0 + wl);
    tile[(wl + 0) * 32 + c] = f2bf(fmaxf(v.x * sc + sh, 0.f));
    tile[(wl + 1) * 32 + c] = f2bf(fmaxf(v.y * sc + sh, 0.f));
    tile[(wl + 2) * 32 + c] = f2bf(fmaxf(v.z * sc + sh, 0.f));
    tile[(wl + 3) * 32 + c] = f2bf(fmaxf(v.w * sc + sh, 0.f));
    __syncthreads();

    int p  = t >> 3;
    int c0 = (t & 7) * 4;
    U2 out = *reinterpret_cast<const U2*>(&tile[p * 32 + c0]);
    *reinterpret_cast<U2*>(
        a1 + ((((size_t)b * 128 + h) * 128 + w0 + p) * 32 + c0)) = out;
}

// ---------------------------------------------------------------------------
// conv3x3 implicit GEMM with v_wmma_f32_16x16x32_bf16.
//   wave tile: M=32 pixels (2 x 16 along W), N=16 out channels, K=32 in ch.
//   block = 4 waves = 64 pixels x 32 channels of one (b, h) row:
//     wave = {pixel half (wv>>1), ntile (wv&1)}.
//   B fragments: register-resident for all 9 taps (loaded from global; the
//   18 KB packed weight array is WGP$/L2 resident and shared by all blocks).
//   MODE 1: + bias, BN2, ReLU -> bf16 NHWC
//   MODE 2: + bias, + residual x -> f32 NCHW
// ---------------------------------------------------------------------------
template <int MODE>
__global__ __launch_bounds__(128)
void conv3x3_wmma_kernel(const unsigned short* __restrict__ act,  // bf16 NHWC
                         const unsigned short* __restrict__ wp,   // packed frags
                         const float* __restrict__ bias,
                         const float* __restrict__ bg, const float* __restrict__ bb,
                         const float* __restrict__ bm, const float* __restrict__ bv,
                         const float* __restrict__ xres,
                         unsigned short* __restrict__ obf,
                         float* __restrict__ of) {
    __shared__ U4 actS4[3 * 66 * 4];   // 3 rows, 66 px (halo), 4x16B/px = 12672 B

    int blk = blockIdx.x;              // B*H*2 = 16384
    int b   = blk >> 8;
    int rem = blk & 255;
    int h   = rem >> 1;
    int w0  = (rem & 1) * 64;
    int tid = threadIdx.x;

    int wv     = tid >> 5;
    int lane   = tid & 31;
    int mtbase = (wv >> 1) * 32;       // pixel half owned by this wave
    int nt     = wv & 1;               // out-channel tile owned by this wave
    int m      = lane & 15;
    int khalf  = lane >> 4;

    // B fragments for this wave's ntile: registers for the whole tile.
    const U4* wp4 = reinterpret_cast<const U4*>(wp);
    AFrag bReg[9];
#pragma unroll
    for (int tap = 0; tap < 9; ++tap) {
        int idx = ((tap * 2 + nt) * 32 + lane) * 2;
        bReg[tap].q[0] = wp4[idx];
        bReg[tap].q[1] = wp4[idx + 1];
    }

    // stage 3 activation rows with 1-pixel halo, zero-padded at borders
    for (int i = tid; i < 792; i += 128) {
        int r  = i / 264;
        int rr = i - r * 264;
        int p  = rr >> 2;
        int q  = rr & 3;
        int hh = h + r - 1;
        int ww = w0 + p - 1;
        U4 v = {0u, 0u, 0u, 0u};
        if ((unsigned)hh < 128u && (unsigned)ww < 128u) {
            v = *reinterpret_cast<const U4*>(
                act + ((((size_t)b * 128 + hh) * 128 + ww) * 32 + q * 8));
        }
        actS4[i] = v;
    }
    __syncthreads();

    v8f acc0 = {0.f, 0.f, 0.f, 0.f, 0.f, 0.f, 0.f, 0.f};
    v8f acc1 = {0.f, 0.f, 0.f, 0.f, 0.f, 0.f, 0.f, 0.f};

#pragma unroll
    for (int kh = 0; kh < 3; ++kh) {
#pragma unroll
        for (int kw = 0; kw < 3; ++kw) {
            int tap = kh * 3 + kw;
            AFrag a0, a1;
            int ab0 = kh * 264 + (mtbase + m + kw) * 4 + khalf * 2;
            int ab1 = ab0 + 64;            // +16 pixels
            a0.q[0] = actS4[ab0];          // K = khalf*16 .. +7
            a0.q[1] = actS4[ab0 + 1];      // K = khalf*16+8 .. +7
            a1.q[0] = actS4[ab1];
            a1.q[1] = actS4[ab1 + 1];
            acc0 = __builtin_amdgcn_wmma_f32_16x16x32_bf16(
                false, a0.v, false, bReg[tap].v, (short)0, acc0, false, false);
            acc1 = __builtin_amdgcn_wmma_f32_16x16x32_bf16(
                false, a1.v, false, bReg[tap].v, (short)0, acc1, false, false);
        }
    }

    // epilogue: C/D layout -> lane = N (out channel), VGPR r = M = pixel
    int nlo = lane & 15;
    int mb  = khalf * 8;
    int ch  = nt * 16 + nlo;
    float bs = bias[ch];
    float sc = 0.f, sh = 0.f;
    if (MODE == 1) {
        sc = bg[ch] * rsqrtf(bv[ch] + EPS);
        sh = bb[ch] - bm[ch] * sc;
    }
#pragma unroll
    for (int mt = 0; mt < 2; ++mt) {
        const v8f& acc = mt ? acc1 : acc0;
        int wpix = w0 + mtbase + mt * 16 + mb;
        if (MODE == 1) {
#pragma unroll
            for (int r = 0; r < 8; ++r) {
                float y = (acc[r] + bs) * sc + sh;
                y = fmaxf(y, 0.f);
                obf[(((size_t)b * 128 + h) * 128 + wpix + r) * 32 + ch] = f2bf(y);
            }
        } else {
            const float* xr = xres + (((size_t)b * 32 + ch) * 128 + h) * 128 + wpix;
            float*       op = of   + (((size_t)b * 32 + ch) * 128 + h) * 128 + wpix;
#pragma unroll
            for (int r = 0; r < 8; ++r) {
                op[r] = acc[r] + bs + xr[r];   // consecutive w -> b128 stores
            }
        }
    }
}

extern "C" void kernel_launch(void* const* d_in, const int* in_sizes, int n_in,
                              void* d_out, int out_size, void* d_ws, size_t ws_size,
                              hipStream_t stream) {
    (void)in_sizes; (void)n_in; (void)out_size; (void)ws_size;
    const float* x    = (const float*)d_in[0];
    const float* b1g  = (const float*)d_in[1];
    const float* b1b  = (const float*)d_in[2];
    const float* b1m  = (const float*)d_in[3];
    const float* b1v  = (const float*)d_in[4];
    const float* w1   = (const float*)d_in[5];
    const float* bi1  = (const float*)d_in[6];
    const float* b2g  = (const float*)d_in[7];
    const float* b2b  = (const float*)d_in[8];
    const float* b2m  = (const float*)d_in[9];
    const float* b2v  = (const float*)d_in[10];
    const float* w2   = (const float*)d_in[11];
    const float* bi2  = (const float*)d_in[12];

    char* ws = (char*)d_ws;
    unsigned short* a1  = (unsigned short*)(ws);
    unsigned short* a2  = (unsigned short*)(ws + (size_t)67108864);
    unsigned short* wp1 = (unsigned short*)(ws + (size_t)134217728);
    unsigned short* wp2 = wp1 + 9216;

    pack_weights_kernel<<<18, 32, 0, stream>>>(w1, wp1);
    pack_weights_kernel<<<18, 32, 0, stream>>>(w2, wp2);

    bn_relu_pack_kernel<<<32768, 256, 0, stream>>>(x, b1g, b1b, b1m, b1v, a1);

    conv3x3_wmma_kernel<1><<<16384, 128, 0, stream>>>(
        a1, wp1, bi1, b2g, b2b, b2m, b2v, nullptr, a2, nullptr);

    conv3x3_wmma_kernel<2><<<16384, 128, 0, stream>>>(
        a2, wp2, bi2, nullptr, nullptr, nullptr, nullptr, x, nullptr, (float*)d_out);
}
